// TurboQuantMSE_68616397520950
// MI455X (gfx1250) — matrix operand order repaired and checked by reading, not compiled, not run on hardware
//
#include <hip/hip_runtime.h>

typedef __attribute__((ext_vector_type(16))) _Float16 v16h;
typedef __attribute__((ext_vector_type(8)))  _Float16 v8h;
typedef __attribute__((ext_vector_type(8)))  float    v8f;
typedef __attribute__((ext_vector_type(4)))  float    v4f;

#define DIM 128
#define NLEV 16
#define TILE_M 16
#define TILES_PER_BLOCK 8
#define NTHREADS 256
#define RSQRT_D 0.088388347648318447f    /* 1/sqrt(128) */
#define SQRT_D  11.313708498984760390f   /* sqrt(128)   */

// Build a 16-bit WMMA A fragment from a row pointer:
// elems 0..7  = 8 contiguous f16 at base   (K = kb*32 + hf*8 .. +7)
// elems 8..15 = 8 contiguous f16 at base+16 (K = kb*32 + 16 + hf*8 .. +7)
__device__ __forceinline__ v16h load_a_frag(const _Float16* base) {
    const v8h lo = *(const v8h*)(base);
    const v8h hi = *(const v8h*)(base + 16);
    return __builtin_shufflevector(lo, hi,
        0, 1, 2, 3, 4, 5, 6, 7, 8, 9, 10, 11, 12, 13, 14, 15);
}

__global__ __launch_bounds__(NTHREADS)
void TurboQuantMSE_kernel(const float* __restrict__ x,
                          const float* __restrict__ Pi,
                          const float* __restrict__ cb,
                          float* __restrict__ out,
                          int nrows)
{
    __shared__ alignas(16) _Float16 xsh[TILE_M * DIM];   // x tile, f16
    __shared__ alignas(16) _Float16 qs[TILE_M * DIM];    // quantized codebook vals
    __shared__ alignas(16) float    outs[TILE_M * DIM];  // staged fp32 output tile
    __shared__ alignas(16) float    partial[TILE_M][16];
    __shared__ alignas(16) float    rn[TILE_M];          // sqrt(d)/max(norm,1e-8)
    __shared__ alignas(16) float    nsc[TILE_M];         // f16-rounded norm / sqrt(d)
    __shared__ _Float16             cbh[NLEV];           // f16 codebook LUT

    const int tid  = threadIdx.x;
    const int wave = tid >> 5;       // 8 waves
    const int lane = tid & 31;
    const int ln   = lane & 15;      // A row M / D col N
    const int hf   = lane >> 4;      // lane half
    const int n0   = wave * 16;      // this wave's N-tile base column

    // ---- codebook thresholds (uniform) + f16 LUT in LDS ----
    float cbv[NLEV];
#pragma unroll
    for (int i = 0; i < NLEV; ++i) cbv[i] = cb[i];
    float thr[NLEV - 1];
#pragma unroll
    for (int i = 0; i < NLEV - 1; ++i) thr[i] = 0.5f * (cbv[i] + cbv[i + 1]);
    if (tid < NLEV) cbh[tid] = (_Float16)cbv[0 /*dummy*/] , cbh[tid] = (_Float16)cb[tid];

    // ---- loop-invariant B tiles in VGPRs ----
    // B layout (32x16 f16, wave32): lane l = col n0+l%16, elem j = K hf*8+j (j<8) / 16+hf*8+j-8 (j>=8).
    v16h B1[4];  // Pi^T: B1[kb][j] = Pi[n0+ln][kb*32+K(j)]
    v16h B2[4];  // Pi  : B2[kb][j] = Pi[kb*32+K(j)][n0+ln]
#pragma unroll
    for (int kb = 0; kb < 4; ++kb) {
#pragma unroll
        for (int j = 0; j < 16; ++j) {
            const int K = kb * 32 + hf * 8 + (j < 8 ? j : j + 8);
            B1[kb][j] = (_Float16)Pi[(n0 + ln) * DIM + K];
            B2[kb][j] = (_Float16)Pi[K * DIM + (n0 + ln)];
        }
    }

    for (int t = 0; t < TILES_PER_BLOCK; ++t) {
        const int row0 = (blockIdx.x * TILES_PER_BLOCK + t) * TILE_M;
        if (row0 + TILE_M > nrows) break;   // block-uniform

        // ---- cooperative load: 8 floats/thread -> norms partial + f16 tile ----
        {
            const v4f* xg = (const v4f*)(x + (size_t)row0 * DIM);
            const v4f a = xg[tid * 2];
            const v4f b = xg[tid * 2 + 1];
            float s = a[0]*a[0] + a[1]*a[1] + a[2]*a[2] + a[3]*a[3]
                    + b[0]*b[0] + b[1]*b[1] + b[2]*b[2] + b[3]*b[3];
            partial[tid >> 4][tid & 15] = s;   // thread's 8 floats = row tid/16, chunk tid%16
            v8h h;
#pragma unroll
            for (int i = 0; i < 4; ++i) { h[i] = (_Float16)a[i]; h[i + 4] = (_Float16)b[i]; }
            *(v8h*)&xsh[tid * 8] = h;
            if (t + 1 < TILES_PER_BLOCK)  // global_prefetch_b8 for next tile
                __builtin_prefetch(x + (size_t)(row0 + TILE_M) * DIM + tid * 8, 0, 3);
        }
        __syncthreads();

        // ---- finish row norms ----
        if (tid < TILE_M) {
            float s = 0.f;
#pragma unroll
            for (int i = 0; i < 16; ++i) s += partial[tid][i];
            const float nrm = __builtin_sqrtf(s);
            rn[tid]  = SQRT_D / fmaxf(nrm, 1e-8f);
            nsc[tid] = (float)(_Float16)nrm * RSQRT_D;  // f16 round-trip, fold 1/sqrt(d)
        }
        __syncthreads();

        // ---- GEMM1: Y = x @ Pi^T (row scale applied to outputs) ----
        v8f acc = {};
#pragma unroll
        for (int kb = 0; kb < 4; ++kb) {
            const v16h A = load_a_frag(&xsh[ln * DIM + kb * 32 + hf * 8]);
            acc = __builtin_amdgcn_wmma_f32_16x16x32_f16(false, A, false, B1[kb],
                                                         (short)0, acc, false, false);
        }

        // ---- quantize: y -> nearest codebook level via binary search ----
        // D layout: lane l, vgpr v -> M = v + hf*8, N = ln.
        const v4f rl = *(const v4f*)&rn[hf * 8];
        const v4f rh = *(const v4f*)&rn[hf * 8 + 4];
#pragma unroll
        for (int v = 0; v < 8; ++v) {
            const float rs = (v < 4) ? rl[v & 3] : rh[v & 3];
            const float y  = acc[v] * rs;
            const bool  b3 = y > thr[7];
            const float ta = b3 ? thr[11] : thr[3];
            const bool  b2 = y > ta;
            const float tb = b3 ? (b2 ? thr[13] : thr[9]) : (b2 ? thr[5] : thr[1]);
            const bool  b1 = y > tb;
            const float tc0 = b1 ? thr[2]  : thr[0];
            const float tc1 = b1 ? thr[6]  : thr[4];
            const float tc2 = b1 ? thr[10] : thr[8];
            const float tc3 = b1 ? thr[14] : thr[12];
            const float tc  = b3 ? (b2 ? tc3 : tc2) : (b2 ? tc1 : tc0);
            const bool  b0  = y > tc;
            const int idx = ((int)b3 << 3) | ((int)b2 << 2) | ((int)b1 << 1) | (int)b0;
            qs[(v + hf * 8) * DIM + n0 + ln] = cbh[idx];   // ds_load_u16 + ds_store_b16
        }
        __syncthreads();

        // ---- GEMM2: Xhat = Q @ Pi ----
        v8f acc2 = {};
#pragma unroll
        for (int kb = 0; kb < 4; ++kb) {
            const v16h A = load_a_frag(&qs[ln * DIM + kb * 32 + hf * 8]);
            acc2 = __builtin_amdgcn_wmma_f32_16x16x32_f16(false, A, false, B2[kb],
                                                          (short)0, acc2, false, false);
        }

        // ---- stage scaled output tile in LDS ----
        const v4f nl = *(const v4f*)&nsc[hf * 8];
        const v4f nh = *(const v4f*)&nsc[hf * 8 + 4];
#pragma unroll
        for (int v = 0; v < 8; ++v) {
            const float ns = (v < 4) ? nl[v & 3] : nh[v & 3];
            outs[(v + hf * 8) * DIM + n0 + ln] = acc2[v] * ns;
        }
        __syncthreads();

        // ---- coalesced fp32 store: 2x global_store_b128 per thread ----
        {
            v4f* og = (v4f*)(out + (size_t)row0 * DIM);
            const v4f* ol = (const v4f*)outs;
            og[tid * 2]     = ol[tid * 2];
            og[tid * 2 + 1] = ol[tid * 2 + 1];
        }
        // no end-of-loop barrier needed: next write to any shared buffer is
        // separated from this tile's reads by >=1 barrier in program order
    }
}

extern "C" void kernel_launch(void* const* d_in, const int* in_sizes, int n_in,
                              void* d_out, int out_size, void* d_ws, size_t ws_size,
                              hipStream_t stream) {
    const float* x  = (const float*)d_in[0];   // (4,8,4096,128) fp32
    const float* Pi = (const float*)d_in[1];   // (128,128) fp32
    const float* cb = (const float*)d_in[2];   // (16,) fp32
    float* out = (float*)d_out;                // (4,8,4096,128) fp32

    const int nrows  = in_sizes[0] / DIM;                   // 131072
    const int blocks = nrows / (TILE_M * TILES_PER_BLOCK);  // 1024
    TurboQuantMSE_kernel<<<blocks, NTHREADS, 0, stream>>>(x, Pi, cb, out, nrows);
}
